// MultiheadAttention_43619687858511
// MI455X (gfx1250) — matrix operand order repaired and checked by reading, not compiled, not run on hardware
//
#include <hip/hip_runtime.h>
#include <hip/hip_bf16.h>
#include <cstddef>

typedef __attribute__((ext_vector_type(16))) _Float16 v16h;
typedef __attribute__((ext_vector_type(8)))  float    v8f;

namespace {
constexpr int BB = 4;
constexpr int NN = 2048;
constexpr int EE = 288;   // embed dim
constexpr int HH = 8;     // heads
constexpr int DD = 36;    // head dim
constexpr int DP = 64;    // padded head dim for score K-loop (2 x K=32)
constexpr int DVP = 48;   // padded head dim for PV column tiles (3 x 16)
constexpr int LDH = HH * DP;  // 512: row stride of padded per-head Q/K storage
}

// ---------------- fragment loaders (contiguous k = 16*(lane>>4) + e in A and B, wave32) -------------

// A-matrix 16x32 f16: lane holds row M = (lane&15); element pair j:
// k = (lane>=16 ? 16 : 0) + 2*j + {0,1}
__device__ __forceinline__ v16h load_a_f32(const float* __restrict__ base,
                                           int ld, int row, int k0, int hi8) {
  const float* r = base + (long)row * ld + k0 + hi8;
  v16h a;
#pragma unroll
  for (int j = 0; j < 8; ++j) {
    int k = j << 1;
    a[2 * j]     = (_Float16)r[k];
    a[2 * j + 1] = (_Float16)r[k + 1];
  }
  return a;
}

__device__ __forceinline__ v16h load_a_f16(const _Float16* __restrict__ base,
                                           int ld, int row, int k0, int hi8) {
  const _Float16* r = base + (long)row * ld + k0 + hi8;
  v16h a;
#pragma unroll
  for (int j = 0; j < 8; ++j) {
    int k = j << 1;
    a[2 * j]     = r[k];
    a[2 * j + 1] = r[k + 1];
  }
  return a;
}

// B-matrix 32x16 f16: lane holds col N = (lane&15); element e: k = hi16 + e.
// Memory holds B^T row-major (contiguous in k for fixed col): one 32B read.
__device__ __forceinline__ v16h load_b_contig(const _Float16* __restrict__ p) {
  v16h b;
#pragma unroll
  for (int e = 0; e < 16; ++e) b[e] = p[e];
  return b;
}

// ---------------- kernel 0: weights -> f16, transposed ----------------------
// Wt[m][x][k] = (f16) W_m[k][x]   (m = 0..3 : Wq, Wk, Wv, Wo)
__global__ void convert_weights_kernel(const float* __restrict__ Wq,
                                       const float* __restrict__ Wk,
                                       const float* __restrict__ Wv,
                                       const float* __restrict__ Wo,
                                       _Float16* __restrict__ Wt) {
  const int m = blockIdx.z;
  const float* W = (m == 0) ? Wq : (m == 1) ? Wk : (m == 2) ? Wv : Wo;
  const int x = blockIdx.y * 16 + threadIdx.y;
  const int k = blockIdx.x * 16 + threadIdx.x;
  Wt[((long)m * EE + x) * EE + k] = (_Float16)W[(long)k * EE + x];
}

// ---------------- kernel 1: fused projection  dst = (X@W + b) * scale -------
// X: (B*N, E) f32 ; Wt: (E, E) f16 TRANSPOSED [n][k] ;
// STORE_VT==false: dst = Q/K padded (B*N, H, DP) f16 (pad pre-zeroed)
// STORE_VT==true : dst = Vt (B, H, DVP, N) f16       (pad pre-zeroed)
template <bool STORE_VT>
__global__ void __launch_bounds__(256)
proj_kernel(const float* __restrict__ X, const _Float16* __restrict__ Wt,
            const float* __restrict__ bias, _Float16* __restrict__ dst,
            float scale) {
  const int lane = threadIdx.x & 31;
  const int wave = threadIdx.x >> 5;
  const int hi8  = (lane >> 4) << 3;
  const int hi16 = (lane >> 4) << 4;
  const int c16  = lane & 15;

  const int mBase = (blockIdx.x * 8 + wave) * 16;   // output row tile
  const int nBase = blockIdx.y * 16;                // output col tile

  v8f acc = {};
  const int rowA = mBase + c16;
  const _Float16* wcol = Wt + (long)(nBase + c16) * EE + hi16;  // B^T row
#pragma unroll
  for (int k0 = 0; k0 < EE; k0 += 32) {
    v16h a  = load_a_f32(X, EE, rowA, k0, hi16);
    v16h bf = load_b_contig(wcol + k0);
    acc = __builtin_amdgcn_wmma_f32_16x16x32_f16(false, a, false, bf,
                                                 (short)0, acc, false, false);
  }

  const int col = nBase + c16;
  const int h = col / DD;
  const int d = col % DD;
  const float bv = bias[col];
#pragma unroll
  for (int j = 0; j < 8; ++j) {
    const int row = mBase + j + hi8;   // C layout: VGPR j -> row j / j+8
    const float v = (acc[j] + bv) * scale;
    if (STORE_VT) {
      const int b = row >> 11;          // N = 2048
      const int n = row & (NN - 1);
      dst[(((long)b * HH + h) * DVP + d) * NN + n] = (_Float16)v;
    } else {
      dst[((long)row * HH + h) * DP + d] = (_Float16)v;
    }
  }
}

// ---------------- kernel 2: flash attention ---------------------------------
// Qp/Kp: (B*N, H, DP) f16 ; Vt: (B, H, DVP, N) f16 ; O: (B*N, E) f16
__global__ void __launch_bounds__(256)
flash_attn_kernel(const _Float16* __restrict__ Qp, const _Float16* __restrict__ Kp,
                  const _Float16* __restrict__ Vt, _Float16* __restrict__ O) {
  __shared__ _Float16 Pl[8][16 * 32];   // per-wave P tile (16 rows x 32 keys)

  const int lane = threadIdx.x & 31;
  const int wave = threadIdx.x >> 5;
  const int hi8  = (lane >> 4) << 3;
  const int hi16 = (lane >> 4) << 4;
  const int c16  = lane & 15;

  const int b = blockIdx.x / HH;
  const int h = blockIdx.x % HH;
  const int nBase = (blockIdx.y * 8 + wave) * 16;   // 16 query rows per wave

  const _Float16* Qb  = Qp + ((long)b * NN * HH + h) * DP;     // + n*LDH
  const _Float16* Kb  = Kp + ((long)b * NN * HH + h) * DP;
  const _Float16* Vtb = Vt + ((long)b * HH + h) * DVP * NN;    // + d*NN + n

  // Q fragments (row = nBase + lane&15), d in [0,32) and [32,64)
  const v16h aQ0 = load_a_f16(Qb, LDH, nBase + c16, 0,  hi16);
  const v16h aQ1 = load_a_f16(Qb, LDH, nBase + c16, 32, hi16);

  float rm[8], rl[8];
  v8f o[3] = {v8f{}, v8f{}, v8f{}};
#pragma unroll
  for (int j = 0; j < 8; ++j) { rm[j] = -1e30f; rl[j] = 0.0f; }

  _Float16* Pw = Pl[wave];

  // ---- K-fragment double buffer: [tile 0/1][d-half 0/1] --------------------
  const _Float16* kr0 = Kb + (long)c16 * LDH + hi16;
  const _Float16* kr1 = Kb + (long)(16 + c16) * LDH + hi16;
  v16h bk00 = load_b_contig(kr0);
  v16h bk01 = load_b_contig(kr0 + 32);
  v16h bk10 = load_b_contig(kr1);
  v16h bk11 = load_b_contig(kr1 + 32);

  for (int m0 = 0; m0 < NN; m0 += 32) {
    // ---- issue all loads for this iteration up front ----------------------
    // V fragments for current block (consumed at the bottom, after softmax)
    v16h bv0 = load_b_contig(Vtb + (long)(0 * 16 + c16) * NN + m0 + hi16);
    v16h bv1 = load_b_contig(Vtb + (long)(1 * 16 + c16) * NN + m0 + hi16);
    v16h bv2 = load_b_contig(Vtb + (long)(2 * 16 + c16) * NN + m0 + hi16);
    // next block's K fragments (clamped to 0 on last iteration: harmless)
    const int mn = (m0 + 32 < NN) ? (m0 + 32) : 0;
    const _Float16* nr0 = Kb + (long)(mn + c16) * LDH + hi16;
    const _Float16* nr1 = Kb + (long)(mn + 16 + c16) * LDH + hi16;
    v16h nk00 = load_b_contig(nr0);
    v16h nk01 = load_b_contig(nr0 + 32);
    v16h nk10 = load_b_contig(nr1);
    v16h nk11 = load_b_contig(nr1 + 32);
    // prefetch next V block into L2
    __builtin_prefetch(Vtb + (long)lane * NN + mn, 0, 3);

    // ---- scores: S = Q . K^T  (two 16x16 tiles covering 32 keys) ----------
    v8f s0 = {}, s1 = {};
    s0 = __builtin_amdgcn_wmma_f32_16x16x32_f16(false, aQ0, false, bk00,
                                                (short)0, s0, false, false);
    s0 = __builtin_amdgcn_wmma_f32_16x16x32_f16(false, aQ1, false, bk01,
                                                (short)0, s0, false, false);
    s1 = __builtin_amdgcn_wmma_f32_16x16x32_f16(false, aQ0, false, bk10,
                                                (short)0, s1, false, false);
    s1 = __builtin_amdgcn_wmma_f32_16x16x32_f16(false, aQ1, false, bk11,
                                                (short)0, s1, false, false);

    // ---- online softmax (rows split: lanes 0-15 -> row j, 16-31 -> j+8) ---
    float alpha[8];
#pragma unroll
    for (int j = 0; j < 8; ++j) {
      float mx = fmaxf(s0[j], s1[j]);
#pragma unroll
      for (int off = 1; off < 16; off <<= 1)
        mx = fmaxf(mx, __shfl_xor(mx, off, 32));
      const float mn2 = fmaxf(rm[j], mx);
      alpha[j] = __expf(rm[j] - mn2);
      const float p0 = __expf(s0[j] - mn2);
      const float p1 = __expf(s1[j] - mn2);
      s0[j] = p0; s1[j] = p1;
      float rs = p0 + p1;
#pragma unroll
      for (int off = 1; off < 16; off <<= 1)
        rs += __shfl_xor(rs, off, 32);
      rl[j] = rl[j] * alpha[j] + rs;
      rm[j] = mn2;
    }
#pragma unroll
    for (int t = 0; t < 3; ++t)
#pragma unroll
      for (int j = 0; j < 8; ++j) o[t][j] *= alpha[j];

    // ---- C-layout -> A-layout transpose of P through LDS ------------------
#pragma unroll
    for (int j = 0; j < 8; ++j) {
      const int row = j + hi8;
      Pw[row * 32 + c16]      = (_Float16)s0[j];
      Pw[row * 32 + 16 + c16] = (_Float16)s1[j];
    }
    __syncthreads();
    v16h aP;
    {
      const _Float16* pr = Pw + c16 * 32 + hi16;   // A row = lane&15
#pragma unroll
      for (int j = 0; j < 8; ++j) {
        const int k = j << 1;
        aP[2 * j]     = pr[k];
        aP[2 * j + 1] = pr[k + 1];
      }
    }
    __syncthreads();

    // ---- O += P . V  (three 16-col tiles of head dim; Vt is d-major) ------
    o[0] = __builtin_amdgcn_wmma_f32_16x16x32_f16(false, aP, false, bv0,
                                                  (short)0, o[0], false, false);
    o[1] = __builtin_amdgcn_wmma_f32_16x16x32_f16(false, aP, false, bv1,
                                                  (short)0, o[1], false, false);
    o[2] = __builtin_amdgcn_wmma_f32_16x16x32_f16(false, aP, false, bv2,
                                                  (short)0, o[2], false, false);

    // rotate K double buffer
    bk00 = nk00; bk01 = nk01; bk10 = nk10; bk11 = nk11;
  }

  // ---- normalize and store (only real d < 36) ------------------------------
  float inv[8];
#pragma unroll
  for (int j = 0; j < 8; ++j) inv[j] = 1.0f / rl[j];
#pragma unroll
  for (int t = 0; t < 3; ++t) {
    const int d = t * 16 + c16;
    if (d < DD) {
#pragma unroll
      for (int j = 0; j < 8; ++j) {
        const int row = nBase + j + hi8;
        O[((long)b * NN + row) * EE + h * DD + d] = (_Float16)(o[t][j] * inv[j]);
      }
    }
  }
}

// ---------------- kernel 3: output projection  Y = O@Wo + bo ----------------
__global__ void __launch_bounds__(256)
out_proj_kernel(const _Float16* __restrict__ O, const _Float16* __restrict__ Wt,
                const float* __restrict__ bias, float* __restrict__ Y) {
  const int lane = threadIdx.x & 31;
  const int wave = threadIdx.x >> 5;
  const int hi8  = (lane >> 4) << 3;
  const int hi16 = (lane >> 4) << 4;
  const int c16  = lane & 15;

  const int mBase = (blockIdx.x * 8 + wave) * 16;
  const int nBase = blockIdx.y * 16;

  v8f acc = {};
  const int rowA = mBase + c16;
  const _Float16* wcol = Wt + (long)(nBase + c16) * EE + hi16;
#pragma unroll
  for (int k0 = 0; k0 < EE; k0 += 32) {
    v16h a  = load_a_f16(O, EE, rowA, k0, hi16);
    v16h bf = load_b_contig(wcol + k0);
    acc = __builtin_amdgcn_wmma_f32_16x16x32_f16(false, a, false, bf,
                                                 (short)0, acc, false, false);
  }

  const int col = nBase + c16;
  const float bv = bias[col];
#pragma unroll
  for (int j = 0; j < 8; ++j) {
    const int row = mBase + j + hi8;
    Y[(long)row * EE + col] = acc[j] + bv;
  }
}

// ---------------- launch ----------------------------------------------------
extern "C" void kernel_launch(void* const* d_in, const int* in_sizes, int n_in,
                              void* d_out, int out_size, void* d_ws, size_t ws_size,
                              hipStream_t stream) {
  const float* q  = (const float*)d_in[0];
  const float* k  = (const float*)d_in[1];
  const float* v  = (const float*)d_in[2];
  const float* Wq = (const float*)d_in[3];
  const float* bq = (const float*)d_in[4];
  const float* Wk = (const float*)d_in[5];
  const float* bk = (const float*)d_in[6];
  const float* Wv = (const float*)d_in[7];
  const float* bv = (const float*)d_in[8];
  const float* Wo = (const float*)d_in[9];
  const float* bo = (const float*)d_in[10];

  _Float16* ws = (_Float16*)d_ws;
  const size_t qkvElems = (size_t)BB * NN * HH * DP;   // 4,194,304 halves
  const size_t vtElems  = (size_t)BB * HH * DVP * NN;  // 3,145,728 halves
  const size_t obElems  = (size_t)BB * NN * EE;        // 2,359,296 halves
  _Float16* Qp  = ws;
  _Float16* Kp  = Qp + qkvElems;
  _Float16* Vtp = Kp + qkvElems;
  _Float16* Ob  = Vtp + vtElems;
  _Float16* Wt  = Ob + obElems;                        // 4 * E * E halves

  // zero Q/K padded storage (d in [36,64)) and Vt pad (d in [36,48))
  hipMemsetAsync(ws, 0, (2 * qkvElems + vtElems) * sizeof(_Float16), stream);

  convert_weights_kernel<<<dim3(EE / 16, EE / 16, 4), dim3(16, 16), 0, stream>>>(
      Wq, Wk, Wv, Wo, Wt);

  const dim3 blk(256);
  const dim3 gProj(BB * NN / 128, EE / 16);   // (64, 18)
  const float scaling = 1.0f / 6.0f;          // D^-0.5, D = 36

  proj_kernel<false><<<gProj, blk, 0, stream>>>(q, Wt + 0 * EE * EE, bq, Qp, scaling);
  proj_kernel<false><<<gProj, blk, 0, stream>>>(k, Wt + 1 * EE * EE, bk, Kp, 1.0f);
  proj_kernel<true ><<<gProj, blk, 0, stream>>>(v, Wt + 2 * EE * EE, bv, Vtp, 1.0f);

  // (B*H, N / (8 waves * 16 rows)) = (32, 16)
  flash_attn_kernel<<<dim3(BB * HH, NN / 128), blk, 0, stream>>>(Qp, Kp, Vtp, Ob);

  out_proj_kernel<<<gProj, blk, 0, stream>>>(Ob, Wt + 3 * EE * EE, bo, (float*)d_out);
}
